// SelfAttention_v1_30794915512749
// MI455X (gfx1250) — matrix operand order repaired and checked by reading, not compiled, not run on hardware
//
#include <hip/hip_runtime.h>
#include <math.h>

// ---------------------------------------------------------------------------
// Types for CDNA5 WMMA (wave32): v_wmma_f32_16x16x32_bf16
// ---------------------------------------------------------------------------
typedef __bf16 bf16;
typedef __attribute__((ext_vector_type(16))) __bf16 v16bf;
typedef __attribute__((ext_vector_type(8)))  __bf16 v8bf;
typedef __attribute__((ext_vector_type(8)))  float  v8f;
typedef __attribute__((ext_vector_type(4)))  unsigned int v4u;
typedef __attribute__((ext_vector_type(8)))  int v8i;
typedef __attribute__((ext_vector_type(4)))  int v4i;

static __device__ __forceinline__ bf16 f2bf(float f) {
    unsigned u = __builtin_bit_cast(unsigned, f);
    unsigned r = u + 0x7FFFu + ((u >> 16) & 1u);   // round-to-nearest-even
    unsigned short h = (unsigned short)(r >> 16);
    return __builtin_bit_cast(bf16, h);
}
static __device__ __forceinline__ float bf2f(bf16 h) {
    unsigned short u = __builtin_bit_cast(unsigned short, h);
    unsigned v = ((unsigned)u) << 16;
    return __builtin_bit_cast(float, v);
}

// ---------------------------------------------------------------------------
// Fragment loaders (global): A 16x32 per ISA 7.12.2, B 32x16 column-contiguous
// ---------------------------------------------------------------------------
static __device__ __forceinline__ v16bf load_fragA(const bf16* __restrict__ base,
                                                   int ld, int row, int k0, int half) {
    const bf16* p = base + (size_t)row * ld + k0 + half * 8;
    v8bf lo = *(const v8bf*)p;
    v8bf hi = *(const v8bf*)(p + 16);
    return __builtin_shufflevector(lo, hi, 0,1,2,3,4,5,6,7,8,9,10,11,12,13,14,15);
}
static __device__ __forceinline__ v16bf load_fragB(const bf16* __restrict__ base,
                                                   int ld, int col, int k0, int half) {
    const bf16* p = base + (size_t)col * ld + k0 + half * 8;
    v8bf lo = *(const v8bf*)p;
    v8bf hi = *(const v8bf*)(p + 16);
    return __builtin_shufflevector(lo, hi, 0,1,2,3,4,5,6,7,8,9,10,11,12,13,14,15);
}

// Fragment loaders (LDS tile, TDM-padded: 32 elems/row + 16B pad -> 40-elem stride)
static __device__ __forceinline__ v16bf lds_fragA(const bf16* base, int row, int half) {
    const bf16* p = base + row * 40 + half * 8;
    v8bf lo = *(const v8bf*)p;
    v8bf hi = *(const v8bf*)(p + 16);
    return __builtin_shufflevector(lo, hi, 0,1,2,3,4,5,6,7,8,9,10,11,12,13,14,15);
}
static __device__ __forceinline__ v16bf lds_fragB(const bf16* base, int row, int half) {
    const bf16* p = base + row * 40 + half * 8;
    v8bf lo = *(const v8bf*)p;
    v8bf hi = *(const v8bf*)(p + 16);
    return __builtin_shufflevector(lo, hi, 0,1,2,3,4,5,6,7,8,9,10,11,12,13,14,15);
}

// C/D tile 16x16 f32: lane l -> n = l&15 ; VGPR r -> m = r + 8*(l>>4)
static __device__ __forceinline__ void store_f32(float* __restrict__ C, int ldc,
                                                 int m0, int n0, int r, int half,
                                                 const v8f& c) {
#pragma unroll
    for (int i = 0; i < 8; ++i)
        C[(size_t)(m0 + 8 * half + i) * ldc + n0 + r] = c[i];
}
static __device__ __forceinline__ void store_bf16(bf16* __restrict__ C, int ldc,
                                                  int m0, int n0, int r, int half,
                                                  const v8f& c) {
#pragma unroll
    for (int i = 0; i < 8; ++i)
        C[(size_t)(m0 + 8 * half + i) * ldc + n0 + r] = f2bf(c[i]);
}
static __device__ __forceinline__ void store_bf16_T(bf16* __restrict__ Ct, int ldct,
                                                    int m0, int n0, int r, int half,
                                                    const v8f& c) {
    v8bf p;
#pragma unroll
    for (int i = 0; i < 8; ++i) p[i] = f2bf(c[i]);
    *(v8bf*)(Ct + (size_t)(n0 + r) * ldct + m0 + 8 * half) = p;
}

static __device__ __forceinline__ v8f wmma_bf16(const v16bf& a, const v16bf& b, const v8f& c) {
    return __builtin_amdgcn_wmma_f32_16x16x32_bf16(false, a, false, b, (short)0, c,
                                                   false, false);
}

// ---------------------------------------------------------------------------
// TDM: issue a 2-D tensor_load_to_lds of (tile_rows x 32) bf16 elements.
// Rows are 64B; pad_enable inserts 16B per 64B stored -> LDS row stride 80B
// (conflict-free fragment reads, 16B alignment preserved).
// D# per CDNA5 ISA ch.8: group0 {count, lds_addr, global_addr, type=2},
// group1 {data_size=2B, pad, tile dims, tensor dims, dim0 stride}.
// Toolchain uses the 6-arg builtin form: (v4u, v8i, v4i, v4i, v8i, i32 cpol).
// ---------------------------------------------------------------------------
static __device__ __forceinline__ void tdm_load_tile(unsigned lds_off,
                                                     const bf16* gptr,
                                                     int tile_rows,
                                                     int row_stride_elems) {
    unsigned long long ga = (unsigned long long)(size_t)gptr;
    v4u g0;
    g0.x = 1u;                                    // count=1 (valid user D#)
    g0.y = lds_off;                               // LDS byte address
    g0.z = (unsigned)(ga & 0xFFFFFFFFu);          // global_addr[95:64]
    g0.w = (unsigned)(ga >> 32) | 0x80000000u;    // global_addr[120:96] | type=2
    v8i g1;
    g1[0] = (1 << 16)        // data_size = 1 -> 2 bytes
          | (1 << 20)        // pad_enable
          | (3 << 22)        // pad_interval: 16 DWORDs (64B)
          | (3 << 25);       // pad_amount:   4 DWORDs (16B)
    g1[1] = 0;                                    // tensor_dim0[15:0] (dim0 = 1<<20)
    g1[2] = 0x10;                                 // tensor_dim0[31:16]; tensor_dim1[15:0]=0
    g1[3] = 0x10 | (32 << 16);                    // tensor_dim1[31:16]=0x10 ; tile_dim0=32
    g1[4] = tile_rows & 0xFFFF;                   // tile_dim1 ; tile_dim2=0
    g1[5] = row_stride_elems;                     // tensor_dim0_stride[31:0]
    g1[6] = 0;                                    // dim0_stride[47:32]; dim1_stride[15:0]
    g1[7] = 0;
    v4i g2 = {0, 0, 0, 0};
    v4i g3 = {0, 0, 0, 0};
    v8i g4 = {0, 0, 0, 0, 0, 0, 0, 0};            // extra group (zero-filled)
    __builtin_amdgcn_tensor_load_to_lds(g0, g1, g2, g3, g4, 0);
}

// ---------------------------------------------------------------------------
// Conversion kernels
// ---------------------------------------------------------------------------
__global__ __launch_bounds__(256) void k_f32_to_bf16(const float* __restrict__ in,
                                                     bf16* __restrict__ out, int n) {
    int i = blockIdx.x * blockDim.x + threadIdx.x;
    if (i < n) out[i] = f2bf(in[i]);
}

// LDS-tiled transpose-convert: out[c][r] = bf16(in[r][c]); coalesced both sides.
__global__ __launch_bounds__(256) void k_f32_to_bf16_T(const float* __restrict__ in,
                                                       bf16* __restrict__ out,
                                                       int rows, int cols) {
    __shared__ bf16 tile[64][65];
    int tc = blockIdx.x * 64;
    int tr = blockIdx.y * 64;
    int tx = threadIdx.x & 63;
    int ty = threadIdx.x >> 6;
#pragma unroll
    for (int rr = ty; rr < 64; rr += 4)
        tile[rr][tx] = f2bf(in[(size_t)(tr + rr) * cols + tc + tx]);
    __syncthreads();
#pragma unroll
    for (int cc = ty; cc < 64; cc += 4)
        out[(size_t)(tc + cc) * rows + tr + tx] = tile[tx][cc];
}

// ---------------------------------------------------------------------------
// TDM-staged dense GEMM: C[M,N] = A[M,K] * B[K,N], B column-contiguous (Bt[N,K]).
// Workgroup = 8 waves (2x4) -> 128x256 output. Per k-step wave0 issues TDM
// loads of A(128x32) and Bt(256x32) into double-buffered LDS; all waves read
// fragments from LDS (ds_load_b128) and run 16 WMMAs each.
// MODE 0: f32 C; 1: bf16 C; 2: bf16 C^T.
// ---------------------------------------------------------------------------
template <int MODE>
__global__ __launch_bounds__(256) void gemm_wmma(const bf16* __restrict__ A, int lda,
                                                 const bf16* __restrict__ Bt, int ldbt,
                                                 void* __restrict__ Cv, int ldc,
                                                 int M, int N, int K) {
    __shared__ __align__(16) bf16 sA[2][128 * 40];   // 2 x 10 KiB
    __shared__ __align__(16) bf16 sB[2][256 * 40];   // 2 x 20 KiB

    int wave = threadIdx.x >> 5;
    int lane = threadIdx.x & 31;
    int r = lane & 15, half = lane >> 4;
    int mwave = (wave >> 2) * 64;                    // 0 or 64 within block tile
    int nwave = (wave & 3) * 64;                     // 0..192 within block tile
    int mblk = blockIdx.y * 128;
    int nblk = blockIdx.x * 256;

    const bf16* Ag = A + (size_t)mblk * lda;
    const bf16* Bg = Bt + (size_t)nblk * ldbt;
    int nk = K >> 5;

    if (wave == 0) {
        tdm_load_tile((unsigned)(size_t)&sA[0][0], Ag, 128, lda);
        tdm_load_tile((unsigned)(size_t)&sB[0][0], Bg, 256, ldbt);
    }

    v8f acc[4][4] = {};
    for (int ks = 0; ks < nk; ++ks) {
        int buf = ks & 1;
        if (wave == 0) {
            if (ks + 1 < nk) {
                int k1 = (ks + 1) << 5;
                tdm_load_tile((unsigned)(size_t)&sA[buf ^ 1][0], Ag + k1, 128, lda);
                tdm_load_tile((unsigned)(size_t)&sB[buf ^ 1][0], Bg + k1, 256, ldbt);
                __builtin_amdgcn_s_wait_tensorcnt(2);   // pair for step ks done
            } else {
                __builtin_amdgcn_s_wait_tensorcnt(0);
            }
        }
        __syncthreads();                               // tile[buf] visible to all

        const bf16* a_t = &sA[buf][0];
        const bf16* b_t = &sB[buf][0];
        v16bf af[4], bfr[4];
#pragma unroll
        for (int i = 0; i < 4; ++i)
            af[i] = lds_fragA(a_t, mwave + 16 * i + r, half);
#pragma unroll
        for (int j = 0; j < 4; ++j)
            bfr[j] = lds_fragB(b_t, nwave + 16 * j + r, half);
#pragma unroll
        for (int i = 0; i < 4; ++i)
#pragma unroll
            for (int j = 0; j < 4; ++j)
                acc[i][j] = wmma_bf16(af[i], bfr[j], acc[i][j]);

        __syncthreads();                               // reads done before overwrite
    }

    int m0 = mblk + mwave, n0 = nblk + nwave;
    if (MODE == 0) {
        float* C = (float*)Cv;
#pragma unroll
        for (int i = 0; i < 4; ++i)
#pragma unroll
            for (int j = 0; j < 4; ++j)
                store_f32(C, ldc, m0 + 16 * i, n0 + 16 * j, r, half, acc[i][j]);
    } else if (MODE == 1) {
        bf16* C = (bf16*)Cv;
#pragma unroll
        for (int i = 0; i < 4; ++i)
#pragma unroll
            for (int j = 0; j < 4; ++j)
                store_bf16(C, ldc, m0 + 16 * i, n0 + 16 * j, r, half, acc[i][j]);
    } else {
        bf16* Ct = (bf16*)Cv;                // Ct is [N x M], ldc = M stride
#pragma unroll
        for (int i = 0; i < 4; ++i)
#pragma unroll
            for (int j = 0; j < 4; ++j)
                store_bf16_T(Ct, ldc, m0 + 16 * i, n0 + 16 * j, r, half, acc[i][j]);
    }
}

// ---------------------------------------------------------------------------
// Causal scores: S[i,j] = Q[i,:] . K[j,:] over lower-triangular 64x64 blocks.
// Direct-global WMMA path (tiles are irregular; working set is L2-resident).
// ---------------------------------------------------------------------------
__global__ __launch_bounds__(256) void scores_wmma(const bf16* __restrict__ Q,
                                                   const bf16* __restrict__ Kb,
                                                   bf16* __restrict__ S,
                                                   int seq, int d) {
    int wave = threadIdx.x >> 5;
    int lane = threadIdx.x & 31;
    int T = seq >> 6;
    int total = T * (T + 1) / 2;
    int w = blockIdx.x * (blockDim.x >> 5) + wave;
    if (w >= total) return;
    int tr = (int)((sqrtf(8.0f * (float)w + 1.0f) - 1.0f) * 0.5f);
    while ((tr + 1) * (tr + 2) / 2 <= w) ++tr;
    while (tr * (tr + 1) / 2 > w) --tr;
    int tc = w - tr * (tr + 1) / 2;
    int m0 = tr << 6, n0 = tc << 6;
    int r = lane & 15, half = lane >> 4;

    v8f acc[4][4] = {};
    for (int k0 = 0; k0 < d; k0 += 32) {
        v16bf af[4], bfr[4];
#pragma unroll
        for (int i = 0; i < 4; ++i)
            af[i] = load_fragA(Q, d, m0 + 16 * i + r, k0, half);
#pragma unroll
        for (int j = 0; j < 4; ++j)
            bfr[j] = load_fragB(Kb, d, n0 + 16 * j + r, k0, half);
#pragma unroll
        for (int i = 0; i < 4; ++i)
#pragma unroll
            for (int j = 0; j < 4; ++j)
                acc[i][j] = wmma_bf16(af[i], bfr[j], acc[i][j]);
    }
#pragma unroll
    for (int i = 0; i < 4; ++i)
#pragma unroll
        for (int j = 0; j < 4; ++j)
            store_bf16(S, seq, m0 + 16 * i, n0 + 16 * j, r, half, acc[i][j]);
}

// ---------------------------------------------------------------------------
// Causal softmax, in-place (vectorized v8bf); chunk-owner threads -> race-free.
// ---------------------------------------------------------------------------
__global__ __launch_bounds__(256) void softmax_causal(bf16* __restrict__ S,
                                                      int cols, float scale) {
    __shared__ float red[256];
    int i = blockIdx.x;
    int tid = threadIdx.x;
    bf16* row = S + (size_t)i * cols;
    int nch = cols >> 3;

    float m = -3.4e38f;
    for (int c = tid; c < nch; c += 256) {
        int j0 = c << 3;
        if (j0 > i) continue;
        v8bf v = *(const v8bf*)(row + j0);
#pragma unroll
        for (int e = 0; e < 8; ++e)
            if (j0 + e <= i) m = fmaxf(m, bf2f(v[e]));
    }
    red[tid] = m; __syncthreads();
    for (int s = 128; s > 0; s >>= 1) {
        if (tid < s) red[tid] = fmaxf(red[tid], red[tid + s]);
        __syncthreads();
    }
    float M = red[0]; __syncthreads();

    float acc = 0.f;
    for (int c = tid; c < nch; c += 256) {
        int j0 = c << 3;
        if (j0 > i) continue;
        v8bf v = *(const v8bf*)(row + j0);
#pragma unroll
        for (int e = 0; e < 8; ++e)
            if (j0 + e <= i) acc += __expf((bf2f(v[e]) - M) * scale);
    }
    red[tid] = acc; __syncthreads();
    for (int s = 128; s > 0; s >>= 1) {
        if (tid < s) red[tid] += red[tid + s];
        __syncthreads();
    }
    float inv = 1.0f / red[0];
    __syncthreads();

    for (int c = tid; c < nch; c += 256) {
        int j0 = c << 3;
        v8bf o;
        if (j0 > i) {
#pragma unroll
            for (int e = 0; e < 8; ++e) o[e] = f2bf(0.0f);
        } else {
            v8bf v = *(const v8bf*)(row + j0);
#pragma unroll
            for (int e = 0; e < 8; ++e)
                o[e] = (j0 + e <= i) ? f2bf(__expf((bf2f(v[e]) - M) * scale) * inv)
                                     : f2bf(0.0f);
        }
        *(v8bf*)(row + j0) = o;
    }
}

// ---------------------------------------------------------------------------
// Host-side orchestration
// ---------------------------------------------------------------------------
extern "C" void kernel_launch(void* const* d_in, const int* in_sizes, int n_in,
                              void* d_out, int out_size, void* d_ws, size_t ws_size,
                              hipStream_t stream) {
    (void)in_sizes; (void)n_in; (void)out_size; (void)ws_size;
    constexpr int SEQ = 4096, D = 2048;
    const float scale = 1.0f / sqrtf((float)D);

    const float* x  = (const float*)d_in[0];
    const float* Wq = (const float*)d_in[1];
    const float* Wk = (const float*)d_in[2];
    const float* Wv = (const float*)d_in[3];
    float* out = (float*)d_out;

    char* ws = (char*)d_ws;
    size_t off = 0;
    bf16* xb  = (bf16*)(ws + off); off += (size_t)SEQ * D * 2;   // x bf16        16 MiB
    bf16* wqt = (bf16*)(ws + off); off += (size_t)D * D * 2;     // Wq^T bf16      8 MiB
    bf16* wkt = (bf16*)(ws + off); off += (size_t)D * D * 2;     // Wk^T bf16      8 MiB
    bf16* wvt = (bf16*)(ws + off); off += (size_t)D * D * 2;     // Wv^T bf16      8 MiB
    bf16* qb  = (bf16*)(ws + off); off += (size_t)SEQ * D * 2;   // Q bf16        16 MiB
    bf16* kb  = (bf16*)(ws + off); off += (size_t)SEQ * D * 2;   // K bf16        16 MiB
    bf16* vt  = (bf16*)(ws + off); off += (size_t)SEQ * D * 2;   // V^T bf16      16 MiB
    bf16* Sb  = (bf16*)(ws + off); off += (size_t)SEQ * SEQ * 2; // scores/P      32 MiB

    const int SD = SEQ * D;

    // 1) precision conversion (+ weight transpose so WMMA B-columns are contiguous)
    k_f32_to_bf16<<<(SD + 255) / 256, 256, 0, stream>>>(x, xb, SD);
    dim3 gT(D / 64, D / 64);
    k_f32_to_bf16_T<<<gT, 256, 0, stream>>>(Wq, wqt, D, D);
    k_f32_to_bf16_T<<<gT, 256, 0, stream>>>(Wk, wkt, D, D);
    k_f32_to_bf16_T<<<gT, 256, 0, stream>>>(Wv, wvt, D, D);

    // 2) projections (TDM-staged GEMMs): Q, K row-major; V stored transposed
    dim3 gProj(D / 256, SEQ / 128);                 // (8, 32)
    gemm_wmma<1><<<gProj, 256, 0, stream>>>(xb, D, wqt, D, qb, D, SEQ, D, D);
    gemm_wmma<1><<<gProj, 256, 0, stream>>>(xb, D, wkt, D, kb, D, SEQ, D, D);
    gemm_wmma<2><<<gProj, 256, 0, stream>>>(xb, D, wvt, D, vt, SEQ, SEQ, D, D);

    // 3) causal scores (lower-triangular 64x64 tiles only)
    int T = SEQ / 64;
    int triWaves = T * (T + 1) / 2;                 // 2080
    scores_wmma<<<(triWaves + 7) / 8, 256, 0, stream>>>(qb, kb, Sb, SEQ, D);

    // 4) causal softmax in place -> P (bf16, zeros above diagonal)
    softmax_causal<<<SEQ, 256, 0, stream>>>(Sb, SEQ, scale);

    // 5) out = P @ V  (TDM-staged GEMM; B supplied as V^T), f32 output
    dim3 gOut(D / 256, SEQ / 128);
    gemm_wmma<0><<<gOut, 256, 0, stream>>>(Sb, SEQ, vt, SEQ, out, D, SEQ, D, SEQ);
}
